// OctonionConv2d_13589276525065
// MI455X (gfx1250) — matrix operand order repaired and checked by reading, not compiled, not run on hardware
//
#include <hip/hip_runtime.h>
#include <hip/hip_bf16.h>
#include <stdint.h>

// ---------------------------------------------------------------------------
// OctonionConv2d == dense conv2d: B=16, Cin=64, Cout=128, H=W=128, 3x3, pad 1.
// Implicit GEMM, bf16 WMMA (f32 accumulate), gfx1250.
// K order: k = (kh*3+kw)*64 + cin (18 slices of 32).
// Block = 128 couts x 64 pixels (one row strip). 8 waves, each wave:
//   16 couts x 64 px = 4 accumulators; one A fetch feeds 4 WMMAs.
// B operand read directly from channel-major bf16 LDS tile (cin padded 64->72
// halfs so the 16 lanes' b128 reads tile all 64 LDS banks conflict-free).
// ---------------------------------------------------------------------------

typedef __attribute__((ext_vector_type(16))) __bf16     v16bf;
typedef __attribute__((ext_vector_type(2)))  __bf16     v2bf;
typedef __attribute__((ext_vector_type(8)))  float      v8f;
typedef __attribute__((ext_vector_type(4)))  unsigned   u32x4;

#define NB    16
#define CIN   64
#define COUT  128
#define HH_   128
#define WW_   128
#define KDIM  576
#define NPX   64          // pixels per block (one strip of a row)
#define XCOLS 66          // NPX + 2 halo
#define CPAD  72          // padded cin (halfs) for bank-conflict-free B reads

__device__ __forceinline__ unsigned short f32_to_bf16_rne(float f) {
    union { float f; unsigned u; } c; c.f = f;
    unsigned u = c.u;
    return (unsigned short)((u + 0x7FFFu + ((u >> 16) & 1u)) >> 16);
}

__device__ __forceinline__ unsigned pack_bf16x2(float lo, float hi) {
#if __has_builtin(__builtin_amdgcn_cvt_pk_bf16_f32)
    v2bf p = __builtin_amdgcn_cvt_pk_bf16_f32(lo, hi);
    return __builtin_bit_cast(unsigned, p);
#else
    return (unsigned)f32_to_bf16_rne(lo) | ((unsigned)f32_to_bf16_rne(hi) << 16);
#endif
}

// -------- weight prep: (16,8,8,8,3,3) f32 -> Wb[cout=128][k=576] bf16 ------
__global__ void octconv_prep_weights(const float* __restrict__ w,
                                     unsigned short* __restrict__ wb) {
    int idx = blockIdx.x * blockDim.x + threadIdx.x;
    if (idx >= COUT * KDIM) return;
    int cout = idx / KDIM;
    int k    = idx - cout * KDIM;
    int khkw = k >> 6;
    int cin  = k & 63;
    int kh = khkw / 3, kw = khkw - kh * 3;
    int i = cout >> 3, p = cout & 7;
    int j = cin  >> 3, q = cin  & 7;
    int src = ((((i * 8 + j) * 8 + p) * 8 + q) * 3 + kh) * 3 + kw;
    wb[idx] = f32_to_bf16_rne(w[src]);
}

// ----------------------------- main kernel ---------------------------------
__launch_bounds__(256, 2)
__global__ void octconv_wmma_kernel(const float* __restrict__ x,
                                    const unsigned short* __restrict__ wb,
                                    const float* __restrict__ bias,
                                    float* __restrict__ out) {
    // f32 halo tile; col stride 67 -> channel stride 201 words (odd, no bank hotspot)
    __shared__ __align__(16) float          xtile[CIN][3][XCOLS + 1];
    // channel-major bf16 tile: [row 3][col 66][cin CPAD]; only cin 0..63 used
    __shared__ __align__(16) unsigned short xbf[3 * XCOLS * CPAD];

    const int tid  = threadIdx.x;
    const int wave = tid >> 5;
    const int lane = tid & 31;

    const int bid = blockIdx.x;
    const int w0  = (bid & 1) << 6;          // 64-pixel strip
    const int h   = (bid >> 1) & 127;
    const int bi  = bid >> 8;

    // ---- phase 1: async global -> LDS f32 halo (coalesced along w) --------
    const unsigned long long xbase = (unsigned long long)(uintptr_t)x;
    for (int idx = tid; idx < CIN * 3 * XCOLS; idx += 256) {
        int cin = idx / (3 * XCOLS);
        int rem = idx - cin * (3 * XCOLS);
        int dr  = rem / XCOLS;
        int col = rem - dr * XCOLS;
        int hr  = h + dr - 1;
        int wc  = w0 + col - 1;
        unsigned lds_off = (unsigned)(uintptr_t)&xtile[cin][dr][col];
        if ((unsigned)hr < 128u && (unsigned)wc < 128u) {
            unsigned long long ga = xbase +
                4ull * ((((unsigned long long)bi * CIN + cin) * HH_ + hr) * WW_ + wc);
            asm volatile("global_load_async_to_lds_b32 %0, %1, off"
                         :: "v"(lds_off), "v"(ga) : "memory");
        } else {
            xtile[cin][dr][col] = 0.0f;
        }
    }
    asm volatile("s_wait_asynccnt 0" ::: "memory");
    __syncthreads();

    // ---- phase 2: transpose + packed bf16 convert into channel-major tile -
    // pair p: cp = p & 31 (cin = 2cp), rc = p >> 5 in [0, 3*66): rc = row*66+col
    unsigned* xbf32 = (unsigned*)xbf;
    for (int p = tid; p < 3 * XCOLS * 32; p += 256) {
        int cp  = p & 31;
        int rc  = p >> 5;
        int row = rc / XCOLS;
        int col = rc - row * XCOLS;
        float f0 = xtile[2 * cp + 0][row][col];
        float f1 = xtile[2 * cp + 1][row][col];
        xbf32[rc * (CPAD / 2) + cp] = pack_bf16x2(f0, f1);
    }
    __syncthreads();

    // ---- phase 3: 18 slices x 4 n-tiles of v_wmma_f32_16x16x32_bf16 -------
    const int coutBase = wave << 4;
    const int m  = lane & 15;
    const int hi = lane >> 4;
    // A: lane needs halfs K = s*32 + hi*16 + [0..15]
    const unsigned short* arow = wb + (unsigned)(coutBase + m) * KDIM + hi * 16;
    // B: per-lane base byte offset (pixel col m -> m cells of CPAD halfs)
    const char* xbfb = (const char*)xbf + (unsigned)m * (CPAD * 2u) + (unsigned)hi * 32u;

    v8f acc0 = {0.f,0.f,0.f,0.f,0.f,0.f,0.f,0.f};
    v8f acc1 = acc0, acc2 = acc0, acc3 = acc0;

    #pragma unroll
    for (int s = 0; s < 18; ++s) {
        const int khkw    = s >> 1;              // compile-time constants
        const int kh      = khkw / 3;
        const int kw      = khkw - kh * 3;
        const int cinbase = (s & 1) << 5;
        // cell (kh, kw + nt*16 + m) * CPAD halfs + cinbase + hi*16 halfs
        const int base    = (kh * XCOLS + kw) * (CPAD * 2) + cinbase * 2;

        union { u32x4 q[2]; v16bf v; } a;
        const u32x4* ap = (const u32x4*)(arow + s * 32);
        a.q[0] = ap[0];
        a.q[1] = ap[1];

        v16bf b0 = *(const v16bf*)(xbfb + base + 0 * 16 * CPAD * 2);
        v16bf b1 = *(const v16bf*)(xbfb + base + 1 * 16 * CPAD * 2);
        v16bf b2 = *(const v16bf*)(xbfb + base + 2 * 16 * CPAD * 2);
        v16bf b3 = *(const v16bf*)(xbfb + base + 3 * 16 * CPAD * 2);

        acc0 = __builtin_amdgcn_wmma_f32_16x16x32_bf16(false, a.v, false, b0,
                                                       (short)0, acc0, false, false);
        acc1 = __builtin_amdgcn_wmma_f32_16x16x32_bf16(false, a.v, false, b1,
                                                       (short)0, acc1, false, false);
        acc2 = __builtin_amdgcn_wmma_f32_16x16x32_bf16(false, a.v, false, b2,
                                                       (short)0, acc2, false, false);
        acc3 = __builtin_amdgcn_wmma_f32_16x16x32_bf16(false, a.v, false, b3,
                                                       (short)0, acc3, false, false);
    }

    // ---- epilogue: C/D layout -> out[bi][cout][h][w0 + nt*16 + m] + bias --
    const int mb = hi * 8;
    #pragma unroll
    for (int r = 0; r < 8; ++r) {
        const int cout = coutBase + mb + r;
        const float bv = bias[cout];
        float* orow = out + (((unsigned long long)bi * COUT + cout) * HH_ + h) * WW_
                          + w0 + m;
        orow[ 0] = acc0[r] + bv;
        orow[16] = acc1[r] + bv;
        orow[32] = acc2[r] + bv;
        orow[48] = acc3[r] + bv;
    }
}

// ---------------------------------------------------------------------------
extern "C" void kernel_launch(void* const* d_in, const int* in_sizes, int n_in,
                              void* d_out, int out_size, void* d_ws, size_t ws_size,
                              hipStream_t stream) {
    const float* x    = (const float*)d_in[0];   // (16, 64, 128, 128)
    const float* w    = (const float*)d_in[1];   // (16, 8, 8, 8, 3, 3)
    const float* bias = (const float*)d_in[2];   // (16, 8)
    float* out        = (float*)d_out;           // (16, 128, 128, 128)
    unsigned short* wb = (unsigned short*)d_ws;  // 128*576 bf16 = 144 KB

    octconv_prep_weights<<<(COUT * KDIM + 255) / 256, 256, 0, stream>>>(w, wb);

    const int nblocks = NB * HH_ * (WW_ / NPX);  // 4096
    octconv_wmma_kernel<<<nblocks, 256, 0, stream>>>(x, wb, bias, out);
}
